// TransformerBlock_26310969655690
// MI455X (gfx1250) — matrix operand order repaired and checked by reading, not compiled, not run on hardware
//
#include <hip/hip_runtime.h>

#define Bn 2
#define Cn 96
#define Hn 256
#define Wn 256
#define HWn 65536
#define HALFn 48
#define NBINSn 9
#define NPATCHn 1024
#define NHPn 32
#define QCn 64
#define PI_F 3.14159265358979323846f
#define CDIV(a,b) (((a)+(b)-1)/(b))

typedef _Float16 f16;
typedef unsigned char u8;
typedef unsigned long long u64;
typedef long long i64;
typedef __attribute__((ext_vector_type(16))) _Float16 v16h;
typedef __attribute__((ext_vector_type(8))) float v8f;

// ---------------- WMMA helpers (gfx1250, wave32) ----------------
__device__ __forceinline__ v8f wmma32(v16h a, v16h b, v8f c) {
  return __builtin_amdgcn_wmma_f32_16x16x32_f16(false, a, false, b, (short)0, c, false, false);
}

// A fragment (16x32 f16): lane m=lane&15 holds row, lane group g picks
// K = 16g..16g+15 (contiguous, matching B)  -> two contiguous 16B runs per lane.
__device__ __forceinline__ v16h ld_afrag(const f16* __restrict__ Arow, int kc, int g) {
  union { float4 f[2]; v16h h; } u;
  u.f[0] = *(const float4*)(Arow + kc + 16 * g);
  u.f[1] = *(const float4*)(Arow + kc + 16 * g + 8);
  return u.h;
}
// B fragment element e -> K = g*16+e (contiguous): 32B straight load
__device__ __forceinline__ v16h ld_16h(const f16* p) {
  union { float4 f[2]; v16h h; } u;
  u.f[0] = *(const float4*)p;
  u.f[1] = *(const float4*)(p + 8);
  return u.h;
}

__device__ __forceinline__ int pixel_of(int n, int pp) {
  int ph = n >> 5, pw = n & 31, py = pp >> 3, px = pp & 7;
  return ((ph << 3) + py) * Wn + (pw << 3) + px;
}
__device__ __forceinline__ float gelu_f(float x) {
  return 0.5f * x * (1.0f + erff(x * 0.70710678118654752f));
}
__device__ __forceinline__ unsigned f2key(float x) {
  unsigned u = __float_as_uint(x);
  return u ^ ((u >> 31) ? 0xFFFFFFFFu : 0x80000000u);
}
__device__ __forceinline__ float key2f(unsigned k) {
  unsigned u = (k >> 31) ? (k ^ 0x80000000u) : ~k;
  return __uint_as_float(u);
}
__device__ __forceinline__ void sobel33(const float* base, int i, int j, float& gx, float& gy) {
  float v[3][3];
#pragma unroll
  for (int di = -1; di <= 1; ++di)
#pragma unroll
    for (int dj = -1; dj <= 1; ++dj) {
      int ii = i + di, jj = j + dj;
      v[di + 1][dj + 1] = (ii >= 0 && ii < Hn && jj >= 0 && jj < Wn) ? base[ii * Wn + jj] : 0.0f;
    }
  gx = (v[0][2] - v[0][0]) + 2.0f * (v[1][2] - v[1][0]) + (v[2][2] - v[2][0]);
  gy = (v[2][0] - v[0][0]) + 2.0f * (v[2][1] - v[0][1]) + (v[2][2] - v[0][2]);
}

// ---------------- generic WMMA GEMMs ----------------
// C(M,N) = A(M,K) * B(K,N). A row-major (K contig). B row-major (N contig),
// staged transposed into LDS (Bs[n][k], padded stride 72) so fragment reads are
// 2x ds_load_b128 per WMMA. Block: 128 thr = 4 waves; tile 64(M) x 32(N); K-unroll 2.
#define BS_STRIDE 72
__global__ void gemm_rm(const f16* __restrict__ A, const f16* __restrict__ B,
                        float* __restrict__ C, int M, int N, int K,
                        i64 sA, i64 sB, i64 sC) {
  __shared__ f16 Bs[32 * BS_STRIDE];
  A += (i64)blockIdx.z * sA; B += (i64)blockIdx.z * sB; C += (i64)blockIdx.z * sC;
  int n0 = blockIdx.x * 32;
  int wave = threadIdx.x >> 5, lane = threadIdx.x & 31;
  int m = lane & 15, g = lane >> 4;
  int row0 = blockIdx.y * 64 + wave * 16;
  bool active = row0 < M;
  const f16* Arow = A + (i64)(row0 + m) * K;
  v8f acc0 = {}, acc1 = {};
  int sq = (threadIdx.x & 3) * 8;   // nn octet within 32-wide N tile
  int skk = threadIdx.x >> 2;       // 0..31 K row
  for (int k0 = 0; k0 < K; k0 += 64) {
    if (k0 + 64 < K) __builtin_prefetch(B + (i64)(k0 + 64 + skk) * N + n0 + sq, 0, 0);
#pragma unroll
    for (int hh = 0; hh < 2; ++hh) {
      int kk = skk + hh * 32;
      f16 tmp[8];
      if (k0 + kk < K) {
        union { float4 f; f16 h[8]; } u;
        u.f = *(const float4*)(B + (i64)(k0 + kk) * N + n0 + sq);
#pragma unroll
        for (int e = 0; e < 8; ++e) tmp[e] = u.h[e];
      } else {
#pragma unroll
        for (int e = 0; e < 8; ++e) tmp[e] = (f16)0.0f;
      }
#pragma unroll
      for (int e = 0; e < 8; ++e) Bs[(sq + e) * BS_STRIDE + kk] = tmp[e];
    }
    __syncthreads();
    if (active) {
#pragma unroll
      for (int ks = 0; ks < 64; ks += 32) {
        if (k0 + ks < K) {
          v16h a  = ld_afrag(Arow, k0 + ks, g);
          v16h b0 = ld_16h(&Bs[m * BS_STRIDE + ks + 16 * g]);
          v16h b1 = ld_16h(&Bs[(16 + m) * BS_STRIDE + ks + 16 * g]);
          acc0 = wmma32(a, b0, acc0);
          acc1 = wmma32(a, b1, acc1);
        }
      }
    }
    __syncthreads();
  }
  if (active) {
#pragma unroll
    for (int i = 0; i < 8; ++i) {
      C[(i64)(row0 + g * 8 + i) * N + n0 + m]      = acc0[i];
      C[(i64)(row0 + g * 8 + i) * N + n0 + 16 + m] = acc1[i];
    }
  }
}

// C(M,N) = A(M,K) * Bt(N,K)^T ; both operands K-contiguous -> all-b128 direct loads.
// Used for attention scores (K=16384, multiple of 64). K-unroll 2.
__global__ void gemm_bt(const f16* __restrict__ A, const f16* __restrict__ Bt,
                        float* __restrict__ C, int M, int N, int K,
                        i64 sA, i64 sB, i64 sC) {
  A += (i64)blockIdx.z * sA; Bt += (i64)blockIdx.z * sB; C += (i64)blockIdx.z * sC;
  int n0 = blockIdx.x * 16;
  int wave = threadIdx.x >> 5, lane = threadIdx.x & 31;
  int m = lane & 15, g = lane >> 4;
  int row0 = blockIdx.y * 64 + wave * 16;
  if (row0 >= M) return;
  const f16* Ar = A + (i64)(row0 + m) * K;
  const f16* Br = Bt + (i64)(n0 + m) * K;
  v8f acc = {};
  for (int k0 = 0; k0 < K; k0 += 64) {
    __builtin_prefetch(Ar + k0 + 256, 0, 0);
    __builtin_prefetch(Br + k0 + 256, 0, 0);
#pragma unroll
    for (int ks = 0; ks < 64; ks += 32) {
      v16h a = ld_afrag(Ar, k0 + ks, g);
      v16h b = ld_16h(Br + k0 + ks + 16 * g);
      acc = wmma32(a, b, acc);
    }
  }
#pragma unroll
  for (int i = 0; i < 8; ++i)
    C[(i64)(row0 + g * 8 + i) * N + n0 + m] = acc[i];
}

// ---------------- elementwise / irregular kernels ----------------
__global__ void k_ln1(const float* __restrict__ x, const float* __restrict__ w,
                      const float* __restrict__ bi, float* __restrict__ y) {
  i64 t = (i64)blockIdx.x * blockDim.x + threadIdx.x;
  if (t >= (i64)Bn * HWn) return;
  int b = (int)(t >> 16), pix = (int)(t & 65535);
  const float* xb = x + ((i64)b * Cn) * HWn + pix;
  float mu = 0.f;
  for (int c = 0; c < Cn; ++c) mu += xb[(i64)c * HWn];
  mu *= (1.0f / Cn);
  float va = 0.f;
  for (int c = 0; c < Cn; ++c) { float d = xb[(i64)c * HWn] - mu; va += d * d; }
  va *= (1.0f / Cn);
  float inv = rsqrtf(va + 1e-5f);
  float* yb = y + ((i64)b * Cn) * HWn + pix;
  for (int c = 0; c < Cn; ++c) yb[(i64)c * HWn] = (xb[(i64)c * HWn] - mu) * inv * w[c] + bi[c];
}

__global__ void k_ln2h(const float* __restrict__ x, const float* __restrict__ w,
                       const float* __restrict__ bi, f16* __restrict__ y) {
  i64 t = (i64)blockIdx.x * blockDim.x + threadIdx.x;
  if (t >= (i64)Bn * HWn) return;
  int b = (int)(t >> 16), pix = (int)(t & 65535);
  const float* xb = x + ((i64)b * Cn) * HWn + pix;
  float mu = 0.f;
  for (int c = 0; c < Cn; ++c) mu += xb[(i64)c * HWn];
  mu *= (1.0f / Cn);
  float va = 0.f;
  for (int c = 0; c < Cn; ++c) { float d = xb[(i64)c * HWn] - mu; va += d * d; }
  va *= (1.0f / Cn);
  float inv = rsqrtf(va + 1e-5f);
  f16* yb = y + ((i64)b * Cn) * HWn + pix;
  for (int c = 0; c < Cn; ++c) yb[(i64)c * HWn] = (f16)((xb[(i64)c * HWn] - mu) * inv * w[c] + bi[c]);
}

__global__ void k_sobelh(const float* __restrict__ y, float* __restrict__ mag, u8* __restrict__ ob) {
  i64 t = (i64)blockIdx.x * blockDim.x + threadIdx.x;
  if (t >= (i64)Bn * HALFn * HWn) return;
  int pix = (int)(t & 65535);
  i64 bc = t >> 16;
  int c = (int)(bc % HALFn), b = (int)(bc / HALFn);
  int i = pix >> 8, j = pix & 255;
  const float* base = y + ((i64)b * Cn + c) * HWn;
  float gx, gy; sobel33(base, i, j, gx, gy);
  float mg = sqrtf(gx * gx + gy * gy + 1e-6f);
  float ang = (atan2f(gy, gx) + PI_F) * ((float)NBINSn / (2.0f * PI_F));
  int bin = ((int)ang) % NBINSn;
  i64 o = ((i64)b * HALFn + c) * HWn + pix;
  mag[o] = mg; ob[o] = (u8)bin;
}

__global__ void k_sv(const float* __restrict__ mag, const u8* __restrict__ ob, float* __restrict__ sv) {
  i64 t = (i64)blockIdx.x * blockDim.x + threadIdx.x;
  if (t >= (i64)Bn * NPATCHn * 64) return;
  int pp = (int)(t & 63);
  i64 bn = t >> 6;
  int b = (int)(bn >> 10), n = (int)(bn & 1023);
  int pix = pixel_of(n, pp);
  float s = 0.f;
  for (int c = 0; c < HALFn; ++c) {
    i64 o = ((i64)b * HALFn + c) * HWn + pix;
    s += mag[o] * ((float)ob[o] + 1.0f);
  }
  sv[t] = s;
}

__global__ void k_hog(const float* __restrict__ mag, const u8* __restrict__ ob, float* __restrict__ hog) {
  __shared__ float bins[NBINSn];
  int bn = blockIdx.x, t = threadIdx.x;
  int b = bn >> 10, n = bn & 1023;
  if (t < NBINSn) bins[t] = 0.f;
  __syncthreads();
  int pix = pixel_of(n, t);
  for (int c = 0; c < HALFn; ++c) {
    i64 o = ((i64)b * HALFn + c) * HWn + pix;
    atomicAdd(&bins[ob[o]], mag[o]);
  }
  __syncthreads();
  if (t == 0) {
    float s = 0.f, h2[NBINSn];
    for (int q = 0; q < NBINSn; ++q) { h2[q] = bins[q] * (1.0f / 3072.0f); s += h2[q]; }
    float inv = 1.0f / (s + 1e-8f);
    for (int q = 0; q < NBINSn; ++q) hog[(i64)bn * NBINSn + q] = h2[q] * inv;
  }
}

__global__ void k_psort(const float* __restrict__ sv, int* __restrict__ perm, int* __restrict__ iperm) {
  __shared__ float kk[64];
  __shared__ int ii[64];
  int bn = blockIdx.x, t = threadIdx.x;
  kk[t] = sv[(i64)bn * 64 + t]; ii[t] = t;
  __syncthreads();
  for (int k = 2; k <= 64; k <<= 1)
    for (int j = k >> 1; j > 0; j >>= 1) {
      int l = t ^ j;
      if (l > t) {
        bool up = ((t & k) == 0);
        float ka = kk[t], kb = kk[l]; int ia = ii[t], ib = ii[l];
        bool gt = (ka > kb) || (ka == kb && ia > ib);
        if (gt == up) { kk[t] = kb; kk[l] = ka; ii[t] = ib; ii[l] = ia; }
      }
      __syncthreads();
    }
  perm[(i64)bn * 64 + t] = ii[t];
  iperm[(i64)bn * 64 + ii[t]] = t;
}

// fused: patch permute + hog-map conv1x1 + bilinear upsample + pack to f16 (c < 48)
__global__ void k_comb(const float* __restrict__ y, const float* __restrict__ hog,
                       const float* __restrict__ wbin, const int* __restrict__ perm,
                       f16* __restrict__ y16) {
  i64 t = (i64)blockIdx.x * blockDim.x + threadIdx.x;
  if (t >= (i64)Bn * HALFn * HWn) return;
  int pix = (int)(t & 65535);
  i64 bc = t >> 16;
  int c = (int)(bc % HALFn), b = (int)(bc / HALFn);
  int i = pix >> 8, j = pix & 255;
  int n = ((i >> 3) << 5) + (j >> 3), pp = ((i & 7) << 3) + (j & 7);
  int spp = perm[((i64)b * NPATCHn + n) * 64 + pp];
  float val = y[((i64)b * Cn + c) * HWn + pixel_of(n, spp)];
  float fx = (j + 0.5f) * 0.125f - 0.5f;
  float fy = (i + 0.5f) * 0.125f - 0.5f;
  int x0 = (int)floorf(fx), y0 = (int)floorf(fy);
  float ax = fx - x0, ay = fy - y0;
  int x0c = x0 < 0 ? 0 : x0, x1c = (x0 + 1 > 31) ? 31 : x0 + 1;
  int y0c = y0 < 0 ? 0 : y0, y1c = (y0 + 1 > 31) ? 31 : y0 + 1;
  const float* wb = wbin + c * NBINSn;
  int ys[2] = {y0c, y1c}, xs[2] = {x0c, x1c};
  float v4[4];
  for (int cy = 0; cy < 2; ++cy)
    for (int cx = 0; cx < 2; ++cx) {
      const float* hp = hog + ((i64)b * NPATCHn + ys[cy] * NHPn + xs[cx]) * NBINSn;
      float s = 0.f;
      for (int q = 0; q < NBINSn; ++q) s += wb[q] * hp[q];
      v4[cy * 2 + cx] = s;
    }
  float hm = (1 - ay) * ((1 - ax) * v4[0] + ax * v4[1]) + ay * ((1 - ax) * v4[2] + ax * v4[3]);
  y16[((i64)b * Cn + c) * HWn + pix] = (f16)(val + hm);
}

__global__ void k_packhi(const float* __restrict__ y, f16* __restrict__ y16) {
  i64 t = (i64)blockIdx.x * blockDim.x + threadIdx.x;
  if (t >= (i64)Bn * HALFn * HWn) return;
  int pix = (int)(t & 65535);
  i64 bc = t >> 16;
  int c = (int)(bc % HALFn) + HALFn, b = (int)(bc / HALFn);
  i64 o = ((i64)b * Cn + c) * HWn + pix;
  y16[o] = (f16)y[o];
}

__global__ void k_cvt(const float* __restrict__ s, f16* __restrict__ d, int n) {
  int t = blockIdx.x * blockDim.x + threadIdx.x;
  if (t < n) d[t] = (f16)s[t];
}

__global__ void k_dw3(const float* __restrict__ src, const float* __restrict__ w, float* __restrict__ dst) {
  i64 t = (i64)blockIdx.x * blockDim.x + threadIdx.x;
  if (t >= (i64)Bn * 480 * HWn) return;
  int pix = (int)(t & 65535);
  i64 bc = t >> 16;
  int ch = (int)(bc % 480), b = (int)(bc / 480);
  int i = pix >> 8, j = pix & 255;
  const float* p = src + ((i64)b * 480 + ch) * HWn;
  const float* wc = w + ch * 9;
  float s = 0.f;
#pragma unroll
  for (int di = -1; di <= 1; ++di) {
    int ii = i + di; if (ii < 0 || ii >= Hn) continue;
#pragma unroll
    for (int dj = -1; dj <= 1; ++dj) {
      int jj = j + dj; if (jj < 0 || jj >= Wn) continue;
      s += wc[(di + 1) * 3 + (dj + 1)] * p[ii * Wn + jj];
    }
  }
  dst[t] = s;
}

__global__ void k_sobelv(const float* __restrict__ qkv, float* __restrict__ g, float* __restrict__ wm) {
  i64 t = (i64)blockIdx.x * blockDim.x + threadIdx.x;
  if (t >= (i64)Bn * HWn) return;
  int b = (int)(t >> 16), pix = (int)(t & 65535);
  int i = pix >> 8, j = pix & 255;
  const float* vb = qkv + ((i64)b * 480 + 384) * HWn;
  float gs = 0.f, ws = 0.f;
  for (int c = 0; c < Cn; ++c) {
    float gx, gy; sobel33(vb + (i64)c * HWn, i, j, gx, gy);
    float mg = sqrtf(gx * gx + gy * gy + 1e-6f);
    float on = (atan2f(gy, gx) + PI_F) * (1.0f / (2.0f * PI_F));
    gs += mg; ws += mg * on;
  }
  g[t] = gs * (1.0f / 96.0f);
  wm[t] = ws;
}

__global__ void k_topk(const float* __restrict__ g, const float* tau, const float* gk,
                       const float* gb, float* __restrict__ alpha) {
  __shared__ float rf[256];
  __shared__ unsigned ru[256];
  int b = blockIdx.x, t = threadIdx.x;
  const float* gp = g + (i64)b * HWn;
  float s = 0.f;
  for (int i = t; i < HWn; i += 256) s += gp[i];
  rf[t] = s; __syncthreads();
  for (int o = 128; o > 0; o >>= 1) { if (t < o) rf[t] += rf[t + o]; __syncthreads(); }
  float gmean = rf[0] * (1.0f / HWn);
  __syncthreads();
  unsigned lo = 0u, hi = 0xFFFFFFFFu;
  while (lo < hi) {
    unsigned d = hi - lo;
    unsigned mid = lo + (d >> 1) + (d & 1u);
    unsigned cnt = 0;
    for (int i = t; i < HWn; i += 256) cnt += (f2key(gp[i]) >= mid) ? 1u : 0u;
    ru[t] = cnt; __syncthreads();
    for (int o = 128; o > 0; o >>= 1) { if (t < o) ru[t] += ru[t + o]; __syncthreads(); }
    unsigned total = ru[0];
    __syncthreads();
    if (total >= 256u) lo = mid; else hi = mid - 1u;
  }
  unsigned tk = lo;
  float sa = 0.f; unsigned ca = 0;
  for (int i = t; i < HWn; i += 256) { unsigned kk = f2key(gp[i]); if (kk > tk) { sa += gp[i]; ca++; } }
  rf[t] = sa; ru[t] = ca; __syncthreads();
  for (int o = 128; o > 0; o >>= 1) { if (t < o) { rf[t] += rf[t + o]; ru[t] += ru[t + o]; } __syncthreads(); }
  if (t == 0) {
    float topsum = rf[0] + (256.0f - (float)ru[0]) * key2f(tk);
    float conf = (topsum * (1.0f / 256.0f)) / (gmean + 1e-6f);
    float a = gk[0] * (conf - tau[0]) + gb[0];
    alpha[b] = 1.0f / (1.0f + expf(-a));
  }
}

// ---------------- bitonic argsort (u64 = key<<32 | idx) ----------------
__global__ void k_sinit(const float* __restrict__ wm, u64* __restrict__ keys) {
  i64 t = (i64)blockIdx.x * blockDim.x + threadIdx.x;
  if (t >= (i64)Bn * HWn) return;
  keys[t] = ((u64)f2key(wm[t]) << 32) | (unsigned)(t & 65535);
}

__global__ void k_bloc(u64* __restrict__ keys) {
  __shared__ u64 sd[2048];
  int blk = blockIdx.x, t = threadIdx.x;
  int chunk = blk & 31;
  i64 base = (i64)blk * 2048;
  sd[t] = keys[base + t]; sd[t + 1024] = keys[base + t + 1024];
  __syncthreads();
  int off = chunk << 11;
  for (int k = 2; k <= 2048; k <<= 1)
    for (int j = k >> 1; j > 0; j >>= 1) {
      int i = ((t & ~(j - 1)) << 1) | (t & (j - 1));
      int l = i + j;
      bool up = (((off + i) & k) == 0);
      u64 A = sd[i], B = sd[l];
      if ((A > B) == up) { sd[i] = B; sd[l] = A; }
      __syncthreads();
    }
  keys[base + t] = sd[t]; keys[base + t + 1024] = sd[t + 1024];
}

__global__ void k_bglob(u64* __restrict__ keys, int k, int j) {
  i64 t = (i64)blockIdx.x * blockDim.x + threadIdx.x;
  if (t >= (i64)Bn * 32768) return;
  int b = (int)(t >> 15), tb = (int)(t & 32767);
  int i = ((tb & ~(j - 1)) << 1) | (tb & (j - 1));
  int l = i + j;
  bool up = ((i & k) == 0);
  u64* kp = keys + ((i64)b << 16);
  u64 A = kp[i], B = kp[l];
  if ((A > B) == up) { kp[i] = B; kp[l] = A; }
}

__global__ void k_bmerge(u64* __restrict__ keys, int k) {
  __shared__ u64 sd[2048];
  int blk = blockIdx.x, t = threadIdx.x;
  int chunk = blk & 31;
  i64 base = (i64)blk * 2048;
  sd[t] = keys[base + t]; sd[t + 1024] = keys[base + t + 1024];
  __syncthreads();
  bool up = (((chunk << 11) & k) == 0);
  for (int j = 1024; j > 0; j >>= 1) {
    int i = ((t & ~(j - 1)) << 1) | (t & (j - 1));
    int l = i + j;
    u64 A = sd[i], B = sd[l];
    if ((A > B) == up) { sd[i] = B; sd[l] = A; }
    __syncthreads();
  }
  keys[base + t] = sd[t]; keys[base + t + 1024] = sd[t + 1024];
}

__global__ void k_extract(const u64* __restrict__ keys, int* __restrict__ idxs) {
  i64 t = (i64)blockIdx.x * blockDim.x + threadIdx.x;
  if (t >= (i64)Bn * HWn) return;
  idxs[t] = (int)(keys[t] & 0xFFFFFFFFu);
}

// ---------------- attention staging ----------------
__global__ void k_rnorm(const float* __restrict__ qkv, const int* __restrict__ idx,
                        float* __restrict__ norms) {
  __shared__ float red[256];
  int c = blockIdx.x, b = blockIdx.y, t = threadIdx.x;
  const float* q1 = qkv + ((i64)b * 480 + c) * HWn;
  const float* k1 = q1 + (i64)96 * HWn;
  const float* q2 = q1 + (i64)192 * HWn;
  const float* k2 = q1 + (i64)288 * HWn;
  const int* ix = idx + (i64)b * HWn;
  float a[16];
  for (int q = 0; q < 16; ++q) a[q] = 0.f;
  for (int j = t; j < HWn; j += 256) {
    int jv = ix[j];
    float v1 = q1[jv], v2 = k1[jv], v3 = q2[jv], v4 = k2[jv];
    int fb = j >> 14, fn = j & 3;
    a[fb] += v1 * v1; a[4 + fb] += v2 * v2; a[8 + fn] += v3 * v3; a[12 + fn] += v4 * v4;
  }
  for (int q = 0; q < 16; ++q) {
    red[t] = a[q]; __syncthreads();
    for (int o = 128; o > 0; o >>= 1) { if (t < o) red[t] += red[t + o]; __syncthreads(); }
    if (t == 0) norms[((i64)b * Cn + c) * 16 + q] = fmaxf(sqrtf(red[0]), 1e-12f);
    __syncthreads();
  }
}

__global__ void k_packattn(const float* __restrict__ qkv, const int* __restrict__ idx,
                           const float* __restrict__ norms,
                           f16* __restrict__ QB, f16* __restrict__ KB, f16* __restrict__ VB,
                           f16* __restrict__ QN, f16* __restrict__ KN, f16* __restrict__ VN) {
  i64 t = (i64)blockIdx.x * blockDim.x + threadIdx.x;
  if (t >= (i64)Bn * Cn * HWn) return;
  int j = (int)(t & 65535);
  i64 bc = t >> 16;
  int c = (int)(bc % Cn), b = (int)(bc / Cn);
  int jv = idx[(i64)b * HWn + j];
  const float* base = qkv + ((i64)b * 480) * HWn;
  float q1 = base[(i64)c * HWn + jv];
  float k1 = base[(i64)(96 + c) * HWn + jv];
  float q2 = base[(i64)(192 + c) * HWn + jv];
  float k2 = base[(i64)(288 + c) * HWn + jv];
  float vv = base[(i64)(384 + c) * HWn + jv];
  const float* nm = norms + ((i64)b * Cn + c) * 16;
  int h = c / 24, ci = c % 24;
  int fb = j >> 14, nb = j & 16383;
  int fn = j & 3, nn = j >> 2;
  i64 zb = ((i64)(b * 4 + h) * 96 + ci * 4 + fb) * 16384 + nb;
  i64 zn = ((i64)(b * 4 + h) * 96 + ci * 4 + fn) * 16384 + nn;
  QB[zb] = (f16)(q1 / nm[fb]);
  KB[zb] = (f16)(k1 / nm[4 + fb]);
  VB[zb] = (f16)vv;
  QN[zn] = (f16)(q2 / nm[8 + fn]);
  KN[zn] = (f16)(k2 / nm[12 + fn]);
  VN[zn] = (f16)vv;
}

__global__ void k_smax(const float* __restrict__ SB, const float* __restrict__ SN,
                       const float* __restrict__ temp, f16* __restrict__ AB, f16* __restrict__ AN) {
  int z = blockIdx.x, which = blockIdx.y, r = threadIdx.x;
  if (r >= 96) return;
  int h = z & 3;
  const float* S = (which ? SN : SB) + (i64)z * 96 * 96 + (i64)r * 96;
  f16* O = (which ? AN : AB) + (i64)z * 96 * 96 + (i64)r * 96;
  float tp = temp[h];
  float sm = 0.f;
  for (int c = 0; c < 96; ++c) sm += expf(S[c] * tp);
  float inv = 1.0f / (sm + 1.0f);
  for (int c = 0; c < 96; ++c) O[c] = (f16)(expf(S[c] * tp) * inv);
}

__global__ void k_unsortp(const float* __restrict__ OBX, const float* __restrict__ ONN,
                          const int* __restrict__ idx, f16* __restrict__ prod) {
  i64 t = (i64)blockIdx.x * blockDim.x + threadIdx.x;
  if (t >= (i64)Bn * Cn * HWn) return;
  int j = (int)(t & 65535);
  i64 bc = t >> 16;
  int c = (int)(bc % Cn), b = (int)(bc / Cn);
  int h = c / 24, ci = c % 24;
  float o1 = OBX[((i64)(b * 4 + h) * 96 + ci * 4 + (j >> 14)) * 16384 + (j & 16383)];
  float o2 = ONN[((i64)(b * 4 + h) * 96 + ci * 4 + (j & 3)) * 16384 + (j >> 2)];
  int jv = idx[(i64)b * HWn + j];
  prod[((i64)b * Cn + c) * HWn + jv] = (f16)(o1 * o2);
}

__global__ void k_resid1(const float* __restrict__ x, const float* __restrict__ att,
                         const int* __restrict__ iperm, const float* __restrict__ alpha,
                         float* __restrict__ xmid) {
  i64 t = (i64)blockIdx.x * blockDim.x + threadIdx.x;
  if (t >= (i64)Bn * Cn * HWn) return;
  int pix = (int)(t & 65535);
  i64 bc = t >> 16;
  int c = (int)(bc % Cn), b = (int)(bc / Cn);
  float val;
  if (c < HALFn) {
    int i = pix >> 8, j = pix & 255;
    int n = ((i >> 3) << 5) + (j >> 3), pp = ((i & 7) << 3) + (j & 7);
    int spp = iperm[((i64)b * NPATCHn + n) * 64 + pp];
    val = att[((i64)b * Cn + c) * HWn + pixel_of(n, spp)];
  } else {
    val = att[t];
  }
  xmid[t] = x[t] + val * alpha[b];
}

__global__ void k_dw5g(const float* __restrict__ zp, const float* __restrict__ w5, float* __restrict__ zg) {
  i64 t = (i64)blockIdx.x * blockDim.x + threadIdx.x;
  if (t >= (i64)Bn * 256 * HWn) return;
  int pix = (int)(t & 65535);
  i64 bc = t >> 16;
  int c = (int)(bc % 256), b = (int)(bc / 256);
  int i = pix >> 8, j = pix & 255;
  const float* p1 = zp + ((i64)b * 512 + c) * HWn;
  const float* p2 = p1 + (i64)256 * HWn;
  const float* w1 = w5 + c * 25;
  const float* w2 = w5 + (c + 256) * 25;
  float a1 = 0.f, a2 = 0.f;
#pragma unroll
  for (int di = -2; di <= 2; ++di) {
    int ii = i + di; if (ii < 0 || ii >= Hn) continue;
#pragma unroll
    for (int dj = -2; dj <= 2; ++dj) {
      int jj = j + dj; if (jj < 0 || jj >= Wn) continue;
      int wi = (di + 2) * 5 + (dj + 2);
      a1 += w1[wi] * p1[ii * Wn + jj];
      a2 += w2[wi] * p2[ii * Wn + jj];
    }
  }
  zg[t] = gelu_f(a1) * a2;
}

__global__ void k_cat(const float* __restrict__ zg, const float* __restrict__ wdec,
                      const float* __restrict__ bdec, const float* __restrict__ sig,
                      f16* __restrict__ cat) {
  i64 t = (i64)blockIdx.x * blockDim.x + threadIdx.x;
  if (t >= (i64)Bn * HWn) return;
  int b = (int)(t >> 16), pix = (int)(t & 65535);
  const float* zb = zg + ((i64)b * 256) * HWn + pix;
  float d = bdec[0];
  for (int c = 0; c < QCn; ++c) d += wdec[c] * zb[(i64)c * HWn];
  d = gelu_f(d);
  f16* cb = cat + ((i64)b * 256) * HWn + pix;
  for (int c = 0; c < 256; ++c) {
    float v = zb[(i64)c * HWn];
    if (c < QCn) v = v + sig[c] * (v - d);
    cb[(i64)c * HWn] = (f16)v;
  }
}

__global__ void k_fin(const float* __restrict__ xmid, const float* __restrict__ zout, float* __restrict__ out) {
  i64 t = (i64)blockIdx.x * blockDim.x + threadIdx.x;
  if (t >= (i64)Bn * Cn * HWn) return;
  out[t] = xmid[t] + zout[t];
}

// ---------------- host ----------------
extern "C" void kernel_launch(void* const* d_in, const int* in_sizes, int n_in,
                              void* d_out, int out_size, void* d_ws, size_t ws_size,
                              hipStream_t stream) {
  (void)in_sizes; (void)n_in; (void)out_size; (void)ws_size;
  const float* x    = (const float*)d_in[0];
  const float* n1w  = (const float*)d_in[1];
  const float* n1b  = (const float*)d_in[2];
  const float* n2w  = (const float*)d_in[3];
  const float* n2b  = (const float*)d_in[4];
  const float* temp = (const float*)d_in[5];
  const float* wqkv = (const float*)d_in[6];
  const float* wdw3 = (const float*)d_in[7];
  const float* wproj= (const float*)d_in[8];
  const float* wbin = (const float*)d_in[9];
  const float* gtau = (const float*)d_in[10];
  const float* gk   = (const float*)d_in[11];
  const float* gb   = (const float*)d_in[12];
  const float* wpin = (const float*)d_in[13];
  const float* wdw5 = (const float*)d_in[14];
  const float* wdec = (const float*)d_in[15];
  const float* bdec = (const float*)d_in[16];
  const float* sig  = (const float*)d_in[17];
  const float* wpout= (const float*)d_in[18];
  float* out = (float*)d_out;

  char* wp = (char*)d_ws;
  auto alloc = [&](size_t bytes) { char* p = wp; wp += (bytes + 255) & ~(size_t)255; return p; };

  float* Y     = (float*)alloc((size_t)Bn * Cn * HWn * 4);
  float* MAG   = (float*)alloc((size_t)Bn * HALFn * HWn * 4);
  u8*    OBIN  = (u8*)   alloc((size_t)Bn * HALFn * HWn);
  float* SV    = (float*)alloc((size_t)Bn * NPATCHn * 64 * 4);
  int*   PERM  = (int*)  alloc((size_t)Bn * NPATCHn * 64 * 4);
  int*   IPERM = (int*)  alloc((size_t)Bn * NPATCHn * 64 * 4);
  float* HOG   = (float*)alloc((size_t)Bn * NPATCHn * NBINSn * 4);
  f16*   W16   = (f16*)  alloc((size_t)512 * 1024);
  f16*   Y16   = (f16*)  alloc((size_t)Bn * Cn * HWn * 2);
  float* QKVP  = (float*)alloc((size_t)Bn * 512 * HWn * 4);   // later reused as ZPRE
  float* QKV   = (float*)alloc((size_t)Bn * 480 * HWn * 4);   // later reused as ZG + CAT16
  float* G     = (float*)alloc((size_t)Bn * HWn * 4);
  float* WM    = (float*)alloc((size_t)Bn * HWn * 4);
  u64*   KEYS  = (u64*)  alloc((size_t)Bn * HWn * 8);
  int*   IDX   = (int*)  alloc((size_t)Bn * HWn * 4);
  float* ALPHA = (float*)alloc(256);
  float* NORMS = (float*)alloc((size_t)Bn * Cn * 16 * 4);
  size_t packB = (size_t)Bn * 4 * 96 * 16384 * 2;
  f16* QB = (f16*)alloc(packB); f16* KB = (f16*)alloc(packB); f16* VB = (f16*)alloc(packB);
  f16* QN = (f16*)alloc(packB); f16* KN = (f16*)alloc(packB); f16* VN = (f16*)alloc(packB);
  float* SBOX = (float*)alloc((size_t)Bn * 4 * 96 * 96 * 4);
  float* SNON = (float*)alloc((size_t)Bn * 4 * 96 * 96 * 4);
  f16*   AB16 = (f16*)  alloc((size_t)Bn * 4 * 96 * 96 * 2);
  f16*   AN16 = (f16*)  alloc((size_t)Bn * 4 * 96 * 96 * 2);
  float* OBOX = (float*)alloc((size_t)Bn * 4 * 96 * 16384 * 4);
  float* ONON = (float*)alloc((size_t)Bn * 4 * 96 * 16384 * 4);
  f16*   PROD = (f16*)  alloc((size_t)Bn * Cn * HWn * 2);
  float* ATT  = (float*)alloc((size_t)Bn * Cn * HWn * 4);     // later reused as ZOUT
  float* XMID = (float*)alloc((size_t)Bn * Cn * HWn * 4);
  f16*   Z16  = (f16*)  alloc((size_t)Bn * Cn * HWn * 2);

  f16* WQKV16  = W16;
  f16* WPROJ16 = W16 + 480 * 96;
  f16* WPIN16  = WPROJ16 + 96 * 96;
  f16* WPOUT16 = WPIN16 + 512 * 96;
  float* ZPRE  = QKVP;
  float* ZG    = QKV;
  f16*   CAT16 = (f16*)((char*)QKV + (size_t)Bn * 256 * HWn * 4);
  float* ZOUT  = ATT;

  const i64 HW = HWn;

  // 1) norm1
  k_ln1<<<512, 256, 0, stream>>>(x, n1w, n1b, Y);
  // 2) sobel + bins on first half
  k_sobelh<<<CDIV(Bn * HALFn * HWn, 256), 256, 0, stream>>>(Y, MAG, OBIN);
  k_sv<<<CDIV(Bn * NPATCHn * 64, 256), 256, 0, stream>>>(MAG, OBIN, SV);
  k_hog<<<Bn * NPATCHn, 64, 0, stream>>>(MAG, OBIN, HOG);
  k_psort<<<Bn * NPATCHn, 64, 0, stream>>>(SV, PERM, IPERM);
  // 3) patch permute + hog-map + pack f16
  k_comb<<<CDIV(Bn * HALFn * HWn, 256), 256, 0, stream>>>(Y, HOG, wbin, PERM, Y16);
  k_packhi<<<CDIV(Bn * HALFn * HWn, 256), 256, 0, stream>>>(Y, Y16);
  // 4) f16 weights
  k_cvt<<<CDIV(480 * 96, 256), 256, 0, stream>>>(wqkv, WQKV16, 480 * 96);
  k_cvt<<<CDIV(96 * 96, 256), 256, 0, stream>>>(wproj, WPROJ16, 96 * 96);
  k_cvt<<<CDIV(512 * 96, 256), 256, 0, stream>>>(wpin, WPIN16, 512 * 96);
  k_cvt<<<CDIV(96 * 256, 256), 256, 0, stream>>>(wpout, WPOUT16, 96 * 256);
  // 5) qkv = dw3(conv1x1) ; WMMA GEMM 480x96x65536 per batch
  gemm_rm<<<dim3(2048, 8, Bn), 128, 0, stream>>>(WQKV16, Y16, QKVP, 480, HWn, 96,
                                                 0, (i64)96 * HW, (i64)480 * HW);
  k_dw3<<<CDIV(Bn * 480 * HWn, 256), 256, 0, stream>>>(QKVP, wdw3, QKV);
  // 6) sobel on v -> gate + sort key
  k_sobelv<<<CDIV(Bn * HWn, 128), 128, 0, stream>>>(QKV, G, WM);
  k_topk<<<Bn, 256, 0, stream>>>(G, gtau, gk, gb, ALPHA);
  // 7) full ascending argsort of wm (stable-equivalent via idx tiebreak)
  k_sinit<<<512, 256, 0, stream>>>(WM, KEYS);
  k_bloc<<<Bn * 32, 1024, 0, stream>>>(KEYS);
  for (int k = 4096; k <= 65536; k <<= 1) {
    for (int j = k >> 1; j >= 2048; j >>= 1)
      k_bglob<<<CDIV(Bn * 32768, 256), 256, 0, stream>>>(KEYS, k, j);
    k_bmerge<<<Bn * 32, 1024, 0, stream>>>(KEYS, k);
  }
  k_extract<<<512, 256, 0, stream>>>(KEYS, IDX);
  // 8) attention staging
  k_rnorm<<<dim3(96, Bn), 256, 0, stream>>>(QKV, IDX, NORMS);
  k_packattn<<<CDIV(Bn * Cn * HWn, 256), 256, 0, stream>>>(QKV, IDX, NORMS, QB, KB, VB, QN, KN, VN);
  // 9) S = q k^T (96x96, K=16384) per (b,head) -- WMMA
  gemm_bt<<<dim3(6, 2, Bn * 4), 128, 0, stream>>>(QB, KB, SBOX, 96, 96, 16384,
                                                  (i64)96 * 16384, (i64)96 * 16384, (i64)96 * 96);
  gemm_bt<<<dim3(6, 2, Bn * 4), 128, 0, stream>>>(QN, KN, SNON, 96, 96, 16384,
                                                  (i64)96 * 16384, (i64)96 * 16384, (i64)96 * 96);
  k_smax<<<dim3(Bn * 4, 2), 128, 0, stream>>>(SBOX, SNON, temp, AB16, AN16);
  // 10) out = attn @ v (96x16384, K=96) -- WMMA
  gemm_rm<<<dim3(512, 2, Bn * 4), 128, 0, stream>>>(AB16, VB, OBOX, 96, 16384, 96,
                                                    (i64)96 * 96, (i64)96 * 16384, (i64)96 * 16384);
  gemm_rm<<<dim3(512, 2, Bn * 4), 128, 0, stream>>>(AN16, VN, ONON, 96, 16384, 96,
                                                    (i64)96 * 96, (i64)96 * 16384, (i64)96 * 16384);
  // 11) unsort + o1*o2 -> proj GEMM
  k_unsortp<<<CDIV(Bn * Cn * HWn, 256), 256, 0, stream>>>(OBOX, ONON, IDX, PROD);
  gemm_rm<<<dim3(2048, 2, Bn), 128, 0, stream>>>(WPROJ16, PROD, ATT, 96, HWn, 96,
                                                 0, (i64)96 * HW, (i64)96 * HW);
  // 12) patch unsort + alpha gate + residual
  k_resid1<<<CDIV(Bn * Cn * HWn, 256), 256, 0, stream>>>(x, ATT, IPERM, ALPHA, XMID);
  // 13) FFN
  k_ln2h<<<512, 256, 0, stream>>>(XMID, n2w, n2b, Z16);
  gemm_rm<<<dim3(2048, 8, Bn), 128, 0, stream>>>(WPIN16, Z16, ZPRE, 512, HWn, 96,
                                                 0, (i64)96 * HW, (i64)512 * HW);
  k_dw5g<<<CDIV(Bn * 256 * HWn, 256), 256, 0, stream>>>(ZPRE, wdw5, ZG);
  k_cat<<<CDIV(Bn * HWn, 128), 128, 0, stream>>>(ZG, wdec, bdec, sig, CAT16);
  gemm_rm<<<dim3(2048, 2, Bn), 128, 0, stream>>>(WPOUT16, CAT16, ZOUT, 96, HWn, 256,
                                                 0, (i64)256 * HW, (i64)96 * HW);
  k_fin<<<CDIV(Bn * Cn * HWn, 256), 256, 0, stream>>>(XMID, ZOUT, out);
}
